// MoleculeSeq2Seq_84885733638663
// MI455X (gfx1250) — compile-verified, matrix operand order checked by reading, not run on hardware
//
#include <hip/hip_runtime.h>
#include <hip/hip_bf16.h>

#define DEV __device__ __forceinline__

typedef __bf16 bf16;
typedef __attribute__((ext_vector_type(16))) __bf16 v16bf;
typedef __attribute__((ext_vector_type(8)))  __bf16 bf16x8;
typedef __attribute__((ext_vector_type(8)))  float  v8f;

static constexpr int BB   = 128;
static constexpr int TT   = 512;
static constexpr int FD   = 256;
static constexpr int ED   = 256;
static constexpr int HD   = 256;
static constexpr int G4   = 4 * HD;          // 1024
static constexpr int KENC = ED + HD + HD;    // 768 : [q | r | q]
static constexpr int ROWS = BB * TT;         // 65536

// ---------------------------------------------------------------- helpers ---
DEV float sigmoidf_(float x) { return 1.0f / (1.0f + __expf(-x)); }

DEV v8f wmma_bf16(v16bf a, v16bf b, v8f c) {
    // D = A(16x32) * B(32x16) + C, fp32 accumulate
    return __builtin_amdgcn_wmma_f32_16x16x32_bf16(false, a, false, b,
                                                   (short)0, c, false, false);
}

// Load a 16x32 fragment from a row-major bf16 matrix where "lane&15" selects
// the row (A: M-row / B: N-column of W stored as W[n][k]) and K is contiguous.
// Per ISA: lanes 0-15 hold K {k0..k0+7, k0+16..k0+23}; lanes 16-31 the other
// two octets.
DEV v16bf frag_row16(const bf16* __restrict__ base, int ld, int k0, int lane) {
    int r    = lane & 15;
    int half = lane >> 4;
    const bf16* p0 = base + r * ld + k0 + half * 8;
    const bf16* p1 = base + r * ld + k0 + 16 + half * 8;
    bf16x8 lo = *(const bf16x8*)p0;
    bf16x8 hi = *(const bf16x8*)p1;
    v16bf f;
#pragma unroll
    for (int j = 0; j < 8; ++j) { f[j] = lo[j]; f[j + 8] = hi[j]; }
    return f;
}

// Encoder A-matrix fragment: virtual xcat = [q(256) | r(256) | q(256)],
// octet-granular region select (regions are multiples of 8).
DEV bf16x8 ld8_xcat(const bf16* __restrict__ q, const bf16* __restrict__ r,
                    int row, int k) {
    const bf16* p;
    if (k < ED)             p = q + row * HD + k;
    else if (k < ED + HD)   p = r + row * ED + (k - ED);
    else                    p = q + row * HD + (k - ED - HD);
    return *(const bf16x8*)p;
}

DEV v16bf frag_a_enc(const bf16* __restrict__ q, const bf16* __restrict__ r,
                     int row0, int k0, int lane) {
    int m    = row0 + (lane & 15);
    int half = lane >> 4;
    bf16x8 lo = ld8_xcat(q, r, m, k0 + half * 8);
    bf16x8 hi = ld8_xcat(q, r, m, k0 + 16 + half * 8);
    v16bf f;
#pragma unroll
    for (int j = 0; j < 8; ++j) { f[j] = lo[j]; f[j + 8] = hi[j]; }
    return f;
}

// ------------------------------------------------------------ prep kernels ---
__global__ void k_cvt_bf16(const float* __restrict__ src, bf16* __restrict__ dst, int n) {
    int i = blockIdx.x * blockDim.x + threadIdx.x;
    if (i < n) dst[i] = (bf16)src[i];
}

__global__ void k_copy_f32(const float* __restrict__ src, float* __restrict__ dst, int n) {
    int i = blockIdx.x * blockDim.x + threadIdx.x;
    if (i < n) dst[i] = src[i];
}

__global__ void k_wcat(const float* __restrict__ Wih, const float* __restrict__ Whh,
                       bf16* __restrict__ Wcat) {
    int i = blockIdx.x * blockDim.x + threadIdx.x;    // n*KENC + k
    if (i >= G4 * KENC) return;
    int n = i / KENC, k = i - n * KENC;
    float v = (k < ED + HD) ? Wih[n * (ED + HD) + k]
                            : Whh[n * HD + (k - (ED + HD))];
    Wcat[i] = (bf16)v;
}

__global__ void k_bias_sum(const float* __restrict__ a, const float* __restrict__ b,
                           float* __restrict__ o, int n) {
    int i = blockIdx.x * blockDim.x + threadIdx.x;
    if (i < n) o[i] = a[i] + b[i];
}

// ---------------------------------------------------------------- embedder ---
// emb[row, e] = sum_f feat[row, f] * Wemb[e, f]   (one 16x16 tile per wave)
__global__ void k_embed(const bf16* __restrict__ featb, const bf16* __restrict__ Wemb,
                        bf16* __restrict__ embb) {
    int w    = blockIdx.x * (blockDim.x >> 5) + (threadIdx.x >> 5);
    int lane = threadIdx.x & 31;
    int rowTile = w >> 4;             // 4096 row tiles
    int colTile = w & 15;             // 16 E tiles
    int row0 = rowTile * 16, n0 = colTile * 16;
    v8f acc;
#pragma unroll
    for (int i = 0; i < 8; ++i) acc[i] = 0.0f;
    const bf16* A  = featb + (long)row0 * FD;
    const bf16* Bw = Wemb + n0 * FD;
#pragma unroll
    for (int k0 = 0; k0 < FD; k0 += 32) {
        v16bf a = frag_row16(A, FD, k0, lane);
        v16bf b = frag_row16(Bw, FD, k0, lane);
        acc = wmma_bf16(a, b, acc);
    }
    int half = lane >> 4, c = lane & 15;
#pragma unroll
    for (int rr = 0; rr < 8; ++rr) {
        int row = row0 + rr + half * 8;
        embb[(long)row * ED + n0 + c] = (bf16)acc[rr];
    }
}

// --------------------------------------------------------- encoder: LSTM ----
// One wave per (16-row, 16-h) tile; computes all 4 gate tiles over K=768.
__global__ void k_enc_lstm(const bf16* __restrict__ qold, const float* __restrict__ cold,
                           const bf16* __restrict__ rbuf,
                           const bf16* __restrict__ Wcat, const float* __restrict__ bias,
                           bf16* __restrict__ qnew, float* __restrict__ cnew) {
    int w    = blockIdx.x * (blockDim.x >> 5) + (threadIdx.x >> 5);
    int lane = threadIdx.x & 31;
    int rowTile = w >> 4;             // 8 row tiles (B=128)
    int hg      = w & 15;             // 16 h groups
    int row0 = rowTile * 16, h0 = hg * 16;
    v8f acc[4];
#pragma unroll
    for (int g = 0; g < 4; ++g)
#pragma unroll
        for (int i = 0; i < 8; ++i) acc[g][i] = 0.0f;

    for (int k0 = 0; k0 < KENC; k0 += 32) {
        v16bf a = frag_a_enc(qold, rbuf, row0, k0, lane);
#pragma unroll
        for (int g = 0; g < 4; ++g) {
            const bf16* Bw = Wcat + (long)(g * HD + h0) * KENC;
            v16bf b = frag_row16(Bw, KENC, k0, lane);
            acc[g] = wmma_bf16(a, b, acc[g]);
        }
    }
    int half = lane >> 4, cidx = lane & 15;
    int h = h0 + cidx;
#pragma unroll
    for (int rr = 0; rr < 8; ++rr) {
        int row = row0 + rr + half * 8;
        float gi = acc[0][rr] + bias[0 * HD + h];
        float gf = acc[1][rr] + bias[1 * HD + h];
        float gg = acc[2][rr] + bias[2 * HD + h];
        float go = acc[3][rr] + bias[3 * HD + h];
        float c2 = sigmoidf_(gf) * cold[row * HD + h] + sigmoidf_(gi) * tanhf(gg);
        cnew[row * HD + h] = c2;
        qnew[row * HD + h] = (bf16)(sigmoidf_(go) * tanhf(c2));
    }
}

// ----------------------------------------------------- encoder: attention ---
// One block per batch row. emb is bf16 (33.5 MB -> L2 resident).
__global__ void k_attn(const bf16* __restrict__ q, const bf16* __restrict__ embb,
                       const bf16* __restrict__ Wattn, const int* __restrict__ lens,
                       bf16* __restrict__ rout) {
    __shared__ float qrow[ED];
    __shared__ float qp[ED];
    __shared__ float sc[TT];
    __shared__ float red[256];
    int b = blockIdx.x;
    int tid = threadIdx.x;            // 256 threads
    qrow[tid] = (float)q[b * HD + tid];
    __syncthreads();
    {   // qp[e] = sum_k q[b,k] * Wattn[e,k]
        float s = 0.f;
        const bf16* wr = Wattn + tid * HD;
        for (int k = 0; k < HD; ++k) s += qrow[k] * (float)wr[k];
        qp[tid] = s;
    }
    __syncthreads();
    int len = lens[b];
    for (int t = tid; t < TT; t += 256) {
        float s = 0.f;
        const bf16* er = embb + ((long)b * TT + t) * ED;
        for (int e = 0; e < ED; ++e) s += (float)er[e] * qp[e];
        sc[t] = (t < len) ? s : -3.0e38f;
    }
    __syncthreads();
    red[tid] = fmaxf(sc[tid], sc[tid + 256]);
    __syncthreads();
    for (int ofs = 128; ofs > 0; ofs >>= 1) {
        if (tid < ofs) red[tid] = fmaxf(red[tid], red[tid + ofs]);
        __syncthreads();
    }
    float mx = red[0];
    __syncthreads();
    float s0 = __expf(sc[tid] - mx);
    float s1 = __expf(sc[tid + 256] - mx);
    sc[tid] = s0; sc[tid + 256] = s1;
    red[tid] = s0 + s1;
    __syncthreads();
    for (int ofs = 128; ofs > 0; ofs >>= 1) {
        if (tid < ofs) red[tid] += red[tid + ofs];
        __syncthreads();
    }
    float inv = 1.0f / red[0];
    float racc = 0.f;
    for (int t = 0; t < TT; ++t)
        racc += sc[t] * (float)embb[((long)b * TT + t) * ED + tid];
    rout[b * ED + tid] = (bf16)(racc * inv);
}

// ------------------------------------------------------------- decoder ------
__global__ void k_hterm(const bf16* __restrict__ ench, const float* __restrict__ Whh,
                        const float* __restrict__ biasdec, float* __restrict__ hterm) {
    int idx = blockIdx.x * blockDim.x + threadIdx.x;    // b*1024 + n
    int b = idx >> 10, n = idx & 1023;
    float s = biasdec[n];
    const float* wr = Whh + n * HD;
    const bf16* hq  = ench + b * HD;
    for (int k = 0; k < HD; ++k) s += (float)hq[k] * wr[k];
    hterm[idx] = s;
}

// One wave per (16-row, 16-h) tile; 4 gate tiles; fused gates + ReLU + W_final
// reduction (cross-lane shfl_xor within the two 16-lane halves).
__global__ void k_dec(const bf16* __restrict__ featb, const bf16* __restrict__ Wdec,
                      const float* __restrict__ hterm, const float* __restrict__ encc,
                      const float* __restrict__ Wfinal, float* __restrict__ partial) {
    int w    = blockIdx.x * (blockDim.x >> 5) + (threadIdx.x >> 5);
    int lane = threadIdx.x & 31;
    int rowTile = w >> 4;             // 4096 row tiles
    int hg      = w & 15;             // 16 h groups
    int row0 = rowTile * 16, h0 = hg * 16;
    int b = row0 / TT;                // all 16 rows share one batch index (T%16==0)
    v8f acc[4];
#pragma unroll
    for (int g = 0; g < 4; ++g)
#pragma unroll
        for (int i = 0; i < 8; ++i) acc[g][i] = 0.0f;

    const bf16* A = featb + (long)row0 * FD;
#pragma unroll
    for (int k0 = 0; k0 < FD; k0 += 32) {
        v16bf a = frag_row16(A, FD, k0, lane);
#pragma unroll
        for (int g = 0; g < 4; ++g) {
            const bf16* Bw = Wdec + (long)(g * HD + h0) * FD;
            v16bf bfrag = frag_row16(Bw, FD, k0, lane);
            acc[g] = wmma_bf16(a, bfrag, acc[g]);
        }
    }
    int half = lane >> 4, cidx = lane & 15;
    int h = h0 + cidx;
    float wf = Wfinal[h];
    float cc = encc[b * HD + h];
    float hi = hterm[b * G4 + 0 * HD + h];
    float hf = hterm[b * G4 + 1 * HD + h];
    float hgv = hterm[b * G4 + 2 * HD + h];
    float ho = hterm[b * G4 + 3 * HD + h];
    float vals[8];
#pragma unroll
    for (int rr = 0; rr < 8; ++rr) {
        float gi = acc[0][rr] + hi;
        float gf = acc[1][rr] + hf;
        float gg = acc[2][rr] + hgv;
        float go = acc[3][rr] + ho;
        float c2 = sigmoidf_(gf) * cc + sigmoidf_(gi) * tanhf(gg);
        float d  = sigmoidf_(go) * tanhf(c2);
        vals[rr] = fmaxf(d, 0.0f) * wf;
    }
#pragma unroll
    for (int rr = 0; rr < 8; ++rr) {
        float v = vals[rr];
        v += __shfl_xor(v, 1);
        v += __shfl_xor(v, 2);
        v += __shfl_xor(v, 4);
        v += __shfl_xor(v, 8);
        if (cidx == 0)
            partial[(long)(row0 + rr + half * 8) * 16 + hg] = v;
    }
}

__global__ void k_final(const float* __restrict__ partial, const int* __restrict__ lens,
                        const float* __restrict__ bfin, float* __restrict__ out) {
    int idx = blockIdx.x * blockDim.x + threadIdx.x;    // 0..ROWS-1
    if (idx >= ROWS) return;
    int b = idx / TT, t = idx - b * TT;
    float s = bfin[0];
#pragma unroll
    for (int g = 0; g < 16; ++g) s += partial[(long)idx * 16 + g];
    out[idx] = (t < lens[b]) ? s : 0.0f;
}

// ------------------------------------------------------------- launcher -----
extern "C" void kernel_launch(void* const* d_in, const int* in_sizes, int n_in,
                              void* d_out, int out_size, void* d_ws, size_t ws_size,
                              hipStream_t stream) {
    (void)in_sizes; (void)n_in; (void)out_size; (void)ws_size;
    const float* features = (const float*)d_in[0];
    const int*   lens     = (const int*)  d_in[1];
    const float* W_embed  = (const float*)d_in[2];
    const float* W_attn   = (const float*)d_in[3];
    const float* W_ih_enc = (const float*)d_in[4];
    const float* W_hh_enc = (const float*)d_in[5];
    const float* b_ih_enc = (const float*)d_in[6];
    const float* b_hh_enc = (const float*)d_in[7];
    const float* W_ih_dec = (const float*)d_in[8];
    const float* W_hh_dec = (const float*)d_in[9];
    const float* b_ih_dec = (const float*)d_in[10];
    const float* b_hh_dec = (const float*)d_in[11];
    const float* W_final  = (const float*)d_in[12];
    const float* b_final  = (const float*)d_in[13];
    const float* q0       = (const float*)d_in[14];
    const float* c0       = (const float*)d_in[15];
    const float* r0       = (const float*)d_in[16];
    float* out = (float*)d_out;

    char* ws = (char*)d_ws;
    size_t off = 0;
    auto carve = [&](size_t bytes) -> char* {
        char* p = ws + off;
        off = (off + bytes + 255) & ~(size_t)255;
        return p;
    };
    bf16*  featb    = (bf16*)carve((size_t)ROWS * FD * 2);
    bf16*  embb     = (bf16*)carve((size_t)ROWS * ED * 2);
    bf16*  Wemb_b   = (bf16*)carve((size_t)ED * FD * 2);
    bf16*  Wattn_b  = (bf16*)carve((size_t)ED * HD * 2);
    bf16*  Wcat_b   = (bf16*)carve((size_t)G4 * KENC * 2);
    bf16*  Wdec_b   = (bf16*)carve((size_t)G4 * FD * 2);
    float* bias_enc = (float*)carve(G4 * 4);
    float* bias_dec = (float*)carve(G4 * 4);
    bf16*  qbuf0    = (bf16*)carve((size_t)BB * HD * 2);
    bf16*  qbuf1    = (bf16*)carve((size_t)BB * HD * 2);
    float* cbuf0    = (float*)carve((size_t)BB * HD * 4);
    float* cbuf1    = (float*)carve((size_t)BB * HD * 4);
    bf16*  rbuf     = (bf16*)carve((size_t)BB * ED * 2);
    float* hterm    = (float*)carve((size_t)BB * G4 * 4);
    float* partial  = (float*)carve((size_t)ROWS * 16 * 4);

    auto cvt = [&](const float* s, bf16* d, int n) {
        k_cvt_bf16<<<(n + 255) / 256, 256, 0, stream>>>(s, d, n);
    };

    // ---- prep: bf16 conversions, fused weights, biases, initial states
    cvt(features, featb, ROWS * FD);
    cvt(W_embed, Wemb_b, ED * FD);
    cvt(W_attn, Wattn_b, ED * HD);
    cvt(W_ih_dec, Wdec_b, G4 * FD);
    cvt(q0, qbuf0, BB * HD);
    cvt(r0, rbuf, BB * ED);
    k_copy_f32<<<(BB * HD + 255) / 256, 256, 0, stream>>>(c0, cbuf0, BB * HD);
    k_wcat<<<(G4 * KENC + 255) / 256, 256, 0, stream>>>(W_ih_enc, W_hh_enc, Wcat_b);
    k_bias_sum<<<4, 256, 0, stream>>>(b_ih_enc, b_hh_enc, bias_enc, G4);
    k_bias_sum<<<4, 256, 0, stream>>>(b_ih_dec, b_hh_dec, bias_dec, G4);

    // ---- embed (WMMA): 65536 tiles, 8 waves/block
    k_embed<<<8192, 256, 0, stream>>>(featb, Wemb_b, embb);

    // ---- encoder scan: 512 sequential steps (LSTM WMMA + attention)
    bf16*  qA = qbuf0; bf16*  qB = qbuf1;
    float* cA = cbuf0; float* cB = cbuf1;
    for (int t = 0; t < TT; ++t) {
        k_enc_lstm<<<32, 128, 0, stream>>>(qA, cA, rbuf, Wcat_b, bias_enc, qB, cB);
        k_attn<<<BB, 256, 0, stream>>>(qB, embb, Wattn_b, lens, rbuf);
        bf16* qt = qA; qA = qB; qB = qt;
        float* ct = cA; cA = cB; cB = ct;
    }
    // final states now in qA / cA

    // ---- decoder (WMMA, fused gates + final projection partials)
    k_hterm<<<(BB * G4) / 256, 256, 0, stream>>>(qA, W_hh_dec, bias_dec, hterm);
    k_dec<<<8192, 256, 0, stream>>>(featb, Wdec_b, hterm, cA, W_final, partial);
    k_final<<<(ROWS + 255) / 256, 256, 0, stream>>>(partial, lens, b_final, out);
}
